// DCNv3_failed_12008728560142
// MI455X (gfx1250) — compile-verified, matrix operand order checked by reading, not run on hardware
//
#include <hip/hip_runtime.h>

// ---------------- problem constants (from reference) ----------------
constexpr int Bc = 2;
constexpr int Cc = 64;      // channels (= out channels)
constexpr int Hc = 192;
constexpr int Wc = 192;
constexpr int HW = Hc * Wc;       // 36864
constexpr int Lc = HW;            // out_H*out_W (stride 1, pad 1)
constexpr int KK = 9;             // 3x3 taps
constexpr int Kdim = Cc * KK;     // 576 main GEMM reduction
constexpr int OFFC = 2 * KK;      // 18 offset channels
constexpr int KRAW = 65 * 9;      // 585 offset-conv reduction
constexpr int KOFF = 608;         // 585 padded to 19*32
constexpr int KX   = 576;         // pure-x part of offset-conv K (ci<64)

typedef __attribute__((ext_vector_type(16))) __bf16 v16bf;
typedef __attribute__((ext_vector_type(8)))  __bf16 v8bf;
typedef __attribute__((ext_vector_type(2)))  __bf16 v2bf;
typedef __attribute__((ext_vector_type(8)))  float  v8f;

// load a 16-bf16 WMMA fragment: two 16B reads at +kb and +kb+16 elements
__device__ inline v16bf frag16(const __bf16* p, int kb) {
    v8bf lo = *(const v8bf*)(p + kb);
    v8bf hi = *(const v8bf*)(p + kb + 16);
    return __builtin_shufflevector(lo, hi,
        0, 1, 2, 3, 4, 5, 6, 7, 8, 9, 10, 11, 12, 13, 14, 15);
}

// ---------------------------------------------------------------
// Kernel 1: weight repacks (both GEMMs), f32 -> bf16.
// ---------------------------------------------------------------
__global__ __launch_bounds__(256)
void dcn_pack(const float* __restrict__ w, const float* __restrict__ wo,
              __bf16* __restrict__ wpk, __bf16* __restrict__ apk) {
    int i = blockIdx.x * 256 + threadIdx.x;
    if (i < Cc * Kdim) {
        int o = i / Kdim;
        int r = i - o * Kdim;
        int t = r >> 6, c = r & 63;
        wpk[i] = (__bf16)w[(o * Cc + c) * KK + t];
    } else {
        int i2 = i - Cc * Kdim;
        if (i2 < 32 * KOFF) {
            int j = i2 / KOFF, k = i2 - j * KOFF;
            apk[i2] = (j < OFFC && k < KRAW) ? (__bf16)wo[j * KRAW + k]
                                             : (__bf16)0.0f;
        }
    }
}

// ---------------------------------------------------------------
// Kernel 2: offset conv as WMMA GEMM  offs[18,L] = A[18,585] @ patch[585,L].
// Meta packed per-k into int4 {base_idx, col_base, rowmask_f32_bits, 0}:
// one unconditional ds_load_b128 per k; masking is value-level cndmask,
// address is med3-clamped (only masked lanes can be OOB).
// ---------------------------------------------------------------
__global__ __launch_bounds__(128)
void dcn_offsets(const float* __restrict__ x, const float* __restrict__ depth,
                 const __bf16* __restrict__ apk, const float* __restrict__ bo,
                 float* __restrict__ offs) {
    __shared__ __attribute__((aligned(16))) __bf16 Bt[2][32 * 40];
    __shared__ int4 km[KX];

    const int tid = threadIdx.x, lane = tid & 31, wv = tid >> 5;
    const int mh = wv & 1, nq = wv >> 1;
    const long gp0 = (long)blockIdx.x * 32;
    const int b  = (int)(gp0 / Lc);
    const int p0 = (int)(gp0 - (long)b * Lc);
    const int oh = p0 / Wc, ow0 = p0 - oh * Wc;   // 32-tile never crosses a row

    const float* xb = x + (long)b * Cc * HW;
    const float* db = depth + (long)b * HW;

    // ---- per-block K metadata (576 entries, ~4.5/thread)
    for (int gk = tid; gk < KX; gk += 128) {
        int ci = gk / 9, r = gk - ci * 9;
        int kh = r / 3, kw = r - kh * 3;
        int ih = oh + kh - 1;
        int ihc = min(max(ih, 0), Hc - 1);
        float rowm = ((unsigned)ih < (unsigned)Hc) ? 1.0f : 0.0f;
        int4 e;
        e.x = ci * HW + ihc * Wc + ow0 + (kw - 1);
        e.y = ow0 + kw - 1;
        e.z = __builtin_bit_cast(int, rowm);
        e.w = 0;
        km[gk] = e;
    }
    __syncthreads();

    v8f acc = {};
    const int m = lane & 15, kb = (lane >> 4) * 8;
    const int n = tid & 31, tg = tid >> 5;    // k is wave-uniform: meta = bcast

#pragma unroll
    for (int s = 0; s < KOFF / 32; ++s) {
        __bf16* bt = Bt[s & 1];
        if (s < 18) {
            // pure-x K steps: unconditional loads, value-level masking
#pragma unroll
            for (int qp = 0; qp < 4; ++qp) {
                int k0  = 2 * (tg + 4 * qp);
                int gk0 = 32 * s + k0;
                int4 m0 = km[gk0];
                int4 m1 = km[gk0 + 1];
                int  i0 = min(max(m0.x + n, 0), Cc * HW - 1);
                int  i1 = min(max(m1.x + n, 0), Cc * HW - 1);
                float k0m = ((unsigned)(m0.y + n) < (unsigned)Wc)
                                ? __builtin_bit_cast(float, m0.z) : 0.0f;
                float k1m = ((unsigned)(m1.y + n) < (unsigned)Wc)
                                ? __builtin_bit_cast(float, m1.z) : 0.0f;
                float v0 = k0m * xb[i0];
                float v1 = k1m * xb[i1];
                v2bf pv = {(__bf16)v0, (__bf16)v1};
                *(v2bf*)&bt[n * 40 + k0] = pv;
            }
        } else {
            // final step: 9 depth taps + 23 zero lanes, still branch-free
#pragma unroll
            for (int qp = 0; qp < 4; ++qp) {
                int k0 = 2 * (tg + 4 * qp);
                float v[2];
#pragma unroll
                for (int e = 0; e < 2; ++e) {
                    int k  = k0 + e;
                    int kk = min(k, 8);
                    int kh = kk / 3, kw = kk - kh * 3;
                    int ih = oh + kh - 1, iw = ow0 + n + kw - 1;
                    float msk = (k < 9) ? 1.0f : 0.0f;
                    msk = ((unsigned)ih < (unsigned)Hc) ? msk : 0.0f;
                    msk = ((unsigned)iw < (unsigned)Wc) ? msk : 0.0f;
                    int idx = min(max(ih, 0), Hc - 1) * Wc
                            + min(max(iw, 0), Wc - 1);
                    v[e] = msk * db[idx];
                }
                v2bf pv = {(__bf16)v[0], (__bf16)v[1]};
                *(v2bf*)&bt[n * 40 + k0] = pv;
            }
        }
        __syncthreads();   // single barrier/step; double buffer allows overlap

        v16bf a  = frag16(apk + (16 * mh + m) * KOFF + 32 * s, kb);
        v16bf bm = frag16(&bt[(nq * 16 + m) * 40], kb);
        acc = __builtin_amdgcn_wmma_f32_16x16x32_bf16(
            false, a, false, bm, (short)0, acc, false, false);
    }

    const int nn = lane & 15, mb = (lane >> 4) * 8;
#pragma unroll
    for (int r = 0; r < 8; ++r) {
        int j = 16 * mh + mb + r;
        if (j < OFFC)
            offs[((long)(b * OFFC + j)) * Lc + p0 + nq * 16 + nn] = acc[r] + bo[j];
    }
}

// ---------------------------------------------------------------
// Kernel 3: fused bilinear sampling + WMMA GEMM.
// Block = 128 threads = 4 wave32; wave wv owns rows 16wv..16wv+15,
// N tile = 32 pixels -> 2 accumulators/wave (A fragment reused).
// Each thread builds adjacent channel pairs: 8 loads (all immediate
// offsets off 2 base addresses) -> packed bf16 -> one ds_store_b32.
// ---------------------------------------------------------------
__global__ __launch_bounds__(128)
void dcn_main(const float* __restrict__ x, const float* __restrict__ offs,
              const __bf16* __restrict__ wpk, float* __restrict__ out) {
    __shared__ __attribute__((aligned(16))) __bf16 Bt[2][32 * 40];
    __shared__ __attribute__((aligned(8)))  int2  mip[KK * 32];   // {top, bottom} base
    __shared__ __attribute__((aligned(16))) float mwv[KK * 32][4];

    const int tid = threadIdx.x, lane = tid & 31, wv = tid >> 5;
    const long gp0 = (long)blockIdx.x * 32;
    const int b  = (int)(gp0 / Lc);
    const int p0 = (int)(gp0 - (long)b * Lc);

    // ---- stage 1: per-(tap,pixel) metadata, padded frame Hp=Wp=194
    for (int it = tid; it < KK * 32; it += 128) {
        int t = it >> 5, n = it & 31;
        int pix = p0 + n;
        int oh = pix / Wc, ow = pix - oh * Wc;
        float offx = offs[((long)(b * OFFC + t)) * Lc + pix];
        float offy = offs[((long)(b * OFFC + KK + t)) * Lc + pix];
        float cx = (float)(oh + t / 3) + offx;          // base(1+oh)+delta(t/3-1)
        float cy = (float)(ow + t % 3) + offy;
        cx = fminf(fmaxf(cx, 0.0f), (float)(Hc + 1));   // [0, Hp-1]
        cy = fminf(fmaxf(cy, 0.0f), (float)(Wc + 1));
        int tlx = (int)floorf(cx); tlx = min(max(tlx, 0), Hc);   // [0, Hp-2]
        int tly = (int)floorf(cy); tly = min(max(tly, 0), Wc);
        float fx1 = cx - (float)tlx, fx0 = 1.0f - fx1;
        float fy1 = cy - (float)tly, fy0 = 1.0f - fy1;
        int uxt = tlx - 1, uxb = tlx;        // unpadded top/bottom rows
        int uyl = tly - 1, uyr = tly;        // unpadded left/right cols
        int cb  = min(max(uyl, 0), Wc - 2);  // column pair base (in-bounds)
        float wc0 = (cb == uyl) ? fy0 : ((cb == uyr) ? fy1 : 0.0f);
        float wc1 = ((cb + 1) == uyl) ? fy0 : (((cb + 1) == uyr) ? fy1 : 0.0f);
        bool  vt = (uxt >= 0) && (uxt < Hc);
        bool  vb = (uxb >= 0) && (uxb < Hc);
        float rwt = vt ? fx0 : 0.0f;
        float rwb = vb ? fx1 : 0.0f;
        int2 ip;
        ip.x = (vt ? uxt : 0) * Wc + cb;
        ip.y = (vb ? uxb : 0) * Wc + cb;
        mip[it] = ip;
        mwv[it][0] = rwt * wc0;
        mwv[it][1] = rwt * wc1;
        mwv[it][2] = rwb * wc0;
        mwv[it][3] = rwb * wc1;
    }
    __syncthreads();

    v8f acc0 = {}, acc1 = {};
    const int m = lane & 15, kb = (lane >> 4) * 8;
    const int n = tid & 31, tg = tid >> 5;      // thread keeps one pixel column
    const float* xg = x + (long)b * Cc * HW + (long)(2 * tg) * HW; // thread base

#pragma unroll
    for (int t = 0; t < KK; ++t) {
        const int mt = t * 32 + n;
        const int2 ip = mip[mt];
        const float w0 = mwv[mt][0], w1 = mwv[mt][1];
        const float w2 = mwv[mt][2], w3 = mwv[mt][3];
#pragma unroll
        for (int h = 0; h < 2; ++h) {
            const int step = 2 * t + h;
            __bf16* bt = Bt[step & 1];
            const float* pt = xg + (unsigned)(h * 32 * HW + ip.x);
            const float* pb = xg + (unsigned)(h * 32 * HW + ip.y);
            // ---- build bf16 B tile: 32 channels x 32 pixels (pairs)
#pragma unroll
            for (int qp = 0; qp < 4; ++qp) {
                const int o0 = qp * 8 * HW;         // imm: qp*1179648 B
                const int o1 = o0 + HW;             // adjacent channel
                float v0 = w0 * pt[o0] + w1 * pt[o0 + 1]
                         + w2 * pb[o0] + w3 * pb[o0 + 1];
                float v1 = w0 * pt[o1] + w1 * pt[o1 + 1]
                         + w2 * pb[o1] + w3 * pb[o1 + 1];
                v2bf pv = {(__bf16)v0, (__bf16)v1};
                *(v2bf*)&bt[n * 40 + 2 * (tg + 4 * qp)] = pv;
            }
            __syncthreads();   // single barrier/step (double-buffered tile)

            v16bf a  = frag16(wpk + (16 * wv + m) * Kdim + t * 64 + h * 32, kb);
            v16bf b0 = frag16(&bt[m * 40], kb);          // pixel half 0
            v16bf b1 = frag16(&bt[(16 + m) * 40], kb);   // pixel half 1
            acc0 = __builtin_amdgcn_wmma_f32_16x16x32_bf16(
                false, a, false, b0, (short)0, acc0, false, false);
            acc1 = __builtin_amdgcn_wmma_f32_16x16x32_bf16(
                false, a, false, b1, (short)0, acc1, false, false);
        }
    }

    // ---- store: D layout  n = lane&15,  M = r + 8*(lane>>4)
    const int nn = lane & 15, mb = (lane >> 4) * 8;
#pragma unroll
    for (int r = 0; r < 8; ++r) {
        int o = 16 * wv + mb + r;
        long ro = ((long)(b * Cc + o)) * Lc + p0;
        out[ro + nn]      = acc0[r];
        out[ro + 16 + nn] = acc1[r];
    }
}

// ---------------------------------------------------------------
extern "C" void kernel_launch(void* const* d_in, const int* in_sizes, int n_in,
                              void* d_out, int out_size, void* d_ws, size_t ws_size,
                              hipStream_t stream) {
    const float* x     = (const float*)d_in[0];
    const float* depth = (const float*)d_in[1];
    const float* wo    = (const float*)d_in[2];
    const float* bo    = (const float*)d_in[3];
    const float* w     = (const float*)d_in[4];
    float* out = (float*)d_out;

    // workspace: [0,73728) wpk bf16 | [73728,112640) apk bf16 | offsets f32
    __bf16* wpk  = (__bf16*)d_ws;
    __bf16* apk  = (__bf16*)((char*)d_ws + (size_t)Cc * Kdim * 2);
    float*  offs = (float*)((char*)d_ws + (size_t)Cc * Kdim * 2 + (size_t)32 * KOFF * 2);

    int packN = Cc * Kdim + 32 * KOFF;                      // 56320
    dcn_pack<<<(packN + 255) / 256, 256, 0, stream>>>(w, wo, wpk, apk);
    dcn_offsets<<<(Bc * Lc) / 32, 128, 0, stream>>>(x, depth, apk, bo, offs);
    dcn_main<<<(Bc * Lc) / 32, 128, 0, stream>>>(x, offs, wpk, out);
}
